// SwinLSTMCell_48713519071753
// MI455X (gfx1250) — matrix operand order repaired and checked by reading, not compiled, not run on hardware
//
#include <hip/hip_runtime.h>
#include <hip/hip_bf16.h>
#include <math.h>

typedef _Float16 h16 __attribute__((ext_vector_type(16)));
typedef _Float16 h8  __attribute__((ext_vector_type(8)));
typedef _Float16 h4  __attribute__((ext_vector_type(4)));
typedef float    f8  __attribute__((ext_vector_type(8)));

#define DEV __device__ __forceinline__
#define WMMA_F16(a, b, c) \
  __builtin_amdgcn_wmma_f32_16x16x32_f16(false, (a), false, (b), (short)0, (c), false, false)

constexpr int Bsz   = 8;
constexpr int IN_CH = 64;
constexpr int HID   = 128;
constexpr int Hh    = 128;
constexpr int Ww    = 128;
constexpr int HW    = Hh * Ww;          // 16384 = 2^14
constexpr int NH    = 4;
constexpr int HD    = 32;
constexpr int GROUPS = 8;
constexpr int CPG   = HID / GROUPS;     // 16
constexpr int KPROJ = IN_CH + HID;      // 192
constexpr float EPS = 1e-5f;
constexpr int NCHUNK = 16;
constexpr int GELEM  = CPG * HW;        // 262144 elems per (b,group)
constexpr int CELEM  = GELEM / NCHUNK;  // 16384

// ---------------- fragment helpers ----------------
DEV h16 combine16(h8 lo, h8 hi) {
  return __builtin_shufflevector(lo, hi, 0,1,2,3,4,5,6,7,8,9,10,11,12,13,14,15);
}

// A fragment (16x32 f16) from row-major [M][K] storage; base = &M[m0][k0], ld in halfs
DEV h16 load_a(const _Float16* base, int ld, int lane) {
  int m = lane & 15;
  int off = (lane >= 16) ? 16 : 0;
  const _Float16* p = base + (size_t)m * ld + off;
  h8 lo = *reinterpret_cast<const h8*>(p);
  h8 hi = *reinterpret_cast<const h8*>(p + 8);
  return combine16(lo, hi);
}

// B fragment (32x16 f16, B[k][n]) from row-major [N][K] storage; base = &W[n0][k0]
DEV h16 load_b_rm(const _Float16* base, int ld, int lane) {
  int n = lane & 15;
  int off = (lane >= 16) ? 16 : 0;
  const _Float16* p = base + (size_t)n * ld + off;
  h8 lo = *reinterpret_cast<const h8*>(p);
  h8 hi = *reinterpret_cast<const h8*>(p + 8);
  return combine16(lo, hi);
}

// B fragment gathering a column block of v (stored [16 tokens][32 d]); rows k>=16 are zero
DEV h16 load_b_vcol(const _Float16* vbase, int d0, int lane) {
  h16 r;
  if (lane < 16) {
    const _Float16* p = vbase + d0 + (lane & 15);
#pragma unroll
    for (int j = 0; j < 16; ++j) r[j] = p[(size_t)j * 32];
  } else {
#pragma unroll
    for (int j = 0; j < 16; ++j) r[j] = (_Float16)0.0f;
  }
  return r;
}

DEV float sigm(float x) { return 1.0f / (1.0f + __expf(-x)); }

// ---------------- 0) weight f32 -> f16 ----------------
__global__ void __launch_bounds__(256) wcvt_kernel(
    const float* __restrict__ wp, const float* __restrict__ wq,
    const float* __restrict__ wo, const float* __restrict__ wg,
    _Float16* __restrict__ hp, _Float16* __restrict__ hq,
    _Float16* __restrict__ ho, _Float16* __restrict__ hg) {
  int i = blockIdx.x * 256 + threadIdx.x;
  if (i < HID * KPROJ)  hp[i] = (_Float16)wp[i];
  if (i < 3 * HID * HID) hq[i] = (_Float16)wq[i];
  if (i < HID * HID)     ho[i] = (_Float16)wo[i];
  if (i < 4 * HID * HID) hg[i] = (_Float16)wg[i];
}

// ---------------- 1) proj GEMM: proj[b,o,p] = Wproj[o,:] . concat(x,h)[b,:,p] + bproj[o]
// M=128(o) K=192 N=16384(p). Per WG: 128x128 output tile, 8 waves each own 16 rows.
__global__ void __launch_bounds__(256) proj_kernel(
    const float* __restrict__ x, const float* __restrict__ h,
    const _Float16* __restrict__ Wp, const float* __restrict__ bp,
    _Float16* __restrict__ projh) {
  __shared__ _Float16 combT[128][32];  // [p][k] transposed activation panel
  int tid = threadIdx.x, lane = tid & 31, wid = tid >> 5;
  int b = blockIdx.x >> 7;
  int p0 = (blockIdx.x & 127) << 7;
  int m0 = wid * 16;
  int hi8 = (lane >= 16) ? 8 : 0;

  __builtin_prefetch(Wp + (size_t)m0 * KPROJ, 0, 3);

  f8 acc[8];
  {
    float bb[8];
#pragma unroll
    for (int r = 0; r < 8; ++r) bb[r] = bp[m0 + r + hi8];
#pragma unroll
    for (int nt = 0; nt < 8; ++nt)
#pragma unroll
      for (int r = 0; r < 8; ++r) acc[nt][r] = bb[r];
  }

  for (int ks = 0; ks < KPROJ / 32; ++ks) {
    int k0 = ks * 32;
    __syncthreads();
    // stage 32(k) x 128(p) panel: float4 loads along p, transpose into LDS as f16
#pragma unroll
    for (int i = 0; i < 4; ++i) {
      int lin4 = tid + i * 256;        // 0..1023 == kk*32 + pq
      int kk = lin4 >> 5;
      int pp4 = (lin4 & 31) << 2;
      int ch = k0 + kk;
      const float* src = (ch < IN_CH) ? (x + ((size_t)b * IN_CH + ch) * HW)
                                      : (h + ((size_t)b * HID + (ch - IN_CH)) * HW);
      float4 v = *reinterpret_cast<const float4*>(src + p0 + pp4);
      combT[pp4 + 0][kk] = (_Float16)v.x;
      combT[pp4 + 1][kk] = (_Float16)v.y;
      combT[pp4 + 2][kk] = (_Float16)v.z;
      combT[pp4 + 3][kk] = (_Float16)v.w;
    }
    __syncthreads();
    h16 afr = load_a(Wp + (size_t)m0 * KPROJ + k0, KPROJ, lane);
#pragma unroll
    for (int nt = 0; nt < 8; ++nt) {
      h16 bfr = load_b_rm(&combT[nt * 16][0], 32, lane);
      acc[nt] = WMMA_F16(afr, bfr, acc[nt]);
    }
  }
#pragma unroll
  for (int nt = 0; nt < 8; ++nt)
#pragma unroll
    for (int r = 0; r < 8; ++r) {
      int o = m0 + r + hi8;
      int p = p0 + nt * 16 + (lane & 15);
      projh[((size_t)b * HID + o) * HW + p] = (_Float16)acc[nt][r];
    }
}

// ---------------- 2) windowed attention: 8 windows per WG, one wave per window
__global__ void __launch_bounds__(256) attn_kernel(
    const _Float16* __restrict__ projh,
    const float* __restrict__ lng, const float* __restrict__ lnb,
    const _Float16* __restrict__ Wq, const float* __restrict__ bq,
    const _Float16* __restrict__ Wo, const float* __restrict__ bo,
    float* __restrict__ attnmap) {
  extern __shared__ char smem[];
  _Float16* Xs = (_Float16*)smem;            // [8][16][128] tokens (then layernormed)
  _Float16* qh = Xs + 8 * 16 * 128;          // [8][16][32]
  _Float16* kh = qh + 8 * 16 * 32;
  _Float16* vh = kh + 8 * 16 * 32;
  float*    sc = (float*)(vh + 8 * 16 * 32); // [8][16][16]
  _Float16* aP = (_Float16*)(sc + 8 * 16 * 16); // [8][16][32] softmax, K-padded w/ zeros
  _Float16* aO = aP + 8 * 16 * 32;           // [8][16][128] per-head outputs

  int tid = threadIdx.x, lane = tid & 31, wid = tid >> 5;

  __builtin_prefetch(Wq, 0, 3);
  __builtin_prefetch(Wo, 0, 3);

  // stage 8 windows: one b64 load (4 contiguous pixels of a window row) per unit
#pragma unroll
  for (int i = 0; i < 16; ++i) {
    int lin = tid + i * 256;   // 0..4095 == ((w*128)+c)*4 + ty
    int ty = lin & 3;
    int c = (lin >> 2) & 127;
    int w = lin >> 9;
    int gw = blockIdx.x * 8 + w;
    int b = gw >> 10;
    int rem = gw & 1023;
    int wy = rem >> 5, wx = rem & 31;
    int y = wy * 4 + ty;
    h4 v = *reinterpret_cast<const h4*>(projh + ((size_t)b * HID + c) * HW + y * Ww + wx * 4);
#pragma unroll
    for (int j = 0; j < 4; ++j) Xs[(w * 16 + ty * 4 + j) * 128 + c] = v[j];
  }
  __syncthreads();

  // LayerNorm per token (128 tokens, one per thread)
  if (tid < 128) {
    _Float16* row = Xs + tid * 128;
    float s = 0.f, q = 0.f;
    for (int c = 0; c < 128; ++c) { float v = (float)row[c]; s += v; q += v * v; }
    float mu = s * (1.0f / 128.0f);
    float var = q * (1.0f / 128.0f) - mu * mu;
    float rinv = rsqrtf(var + EPS);
    for (int c = 0; c < 128; ++c) {
      float v = ((float)row[c] - mu) * rinv * lng[c] + lnb[c];
      row[c] = (_Float16)v;
    }
  }
  __syncthreads();

  const int w = wid;
  int hi8 = (lane >= 16) ? 8 : 0;
  h16 afr[4];
#pragma unroll
  for (int kc = 0; kc < 4; ++kc) afr[kc] = load_a(Xs + w * 16 * 128 + kc * 32, 128, lane);

  const float qscale = 0.17677669529663687f;  // HD^-0.5

  for (int hd = 0; hd < NH; ++hd) {
    // q/k/v for this head (two 16-wide N tiles each), WMMA over K=128
#pragma unroll
    for (int part = 0; part < 3; ++part) {
      _Float16* dst = (part == 0 ? qh : part == 1 ? kh : vh) + w * 16 * 32;
#pragma unroll
      for (int nt = 0; nt < 2; ++nt) {
        int n0 = part * HID + hd * HD + nt * 16;
        f8 acc;
        float bv = bq[n0 + (lane & 15)];
#pragma unroll
        for (int r = 0; r < 8; ++r) acc[r] = bv;
#pragma unroll
        for (int ko = 0; ko < 4; ++ko) {
          h16 bfr = load_b_rm(Wq + (size_t)n0 * HID + ko * 32, HID, lane);
          acc = WMMA_F16(afr[ko], bfr, acc);
        }
#pragma unroll
        for (int r = 0; r < 8; ++r) {
          float v = acc[r];
          if (part == 0) v *= qscale;
          dst[(r + hi8) * 32 + nt * 16 + (lane & 15)] = (_Float16)v;
        }
      }
    }
    __syncthreads();

    // scores = q @ k^T : single 16x16x32 WMMA (K = HD = 32)
    {
      h16 aq = load_a(qh + w * 16 * 32, 32, lane);
      h16 bk = load_b_rm(kh + w * 16 * 32, 32, lane);
      f8 s;
#pragma unroll
      for (int r = 0; r < 8; ++r) s[r] = 0.f;
      s = WMMA_F16(aq, bk, s);
#pragma unroll
      for (int r = 0; r < 8; ++r) sc[(w * 16 + r + hi8) * 16 + (lane & 15)] = s[r];
    }
    __syncthreads();

    // softmax over 16 keys per row (128 rows total)
    if (tid < 128) {
      float* srow = sc + tid * 16;
      float mx = srow[0];
#pragma unroll
      for (int j = 1; j < 16; ++j) mx = fmaxf(mx, srow[j]);
      float e[16]; float sum = 0.f;
#pragma unroll
      for (int j = 0; j < 16; ++j) { e[j] = __expf(srow[j] - mx); sum += e[j]; }
      float inv = 1.0f / sum;
      _Float16* prow = aP + tid * 32;
#pragma unroll
      for (int j = 0; j < 16; ++j) prow[j] = (_Float16)(e[j] * inv);
#pragma unroll
      for (int j = 16; j < 32; ++j) prow[j] = (_Float16)0.0f;  // zero K pad
    }
    __syncthreads();

    // out_head = attn @ v : K padded to 32 with zeros
    {
      h16 ap = load_a(aP + w * 16 * 32, 32, lane);
#pragma unroll
      for (int nt = 0; nt < 2; ++nt) {
        h16 bv = load_b_vcol(vh + w * 16 * 32, nt * 16, lane);
        f8 o;
#pragma unroll
        for (int r = 0; r < 8; ++r) o[r] = 0.f;
        o = WMMA_F16(ap, bv, o);
#pragma unroll
        for (int r = 0; r < 8; ++r)
          aO[(w * 16 + r + hi8) * 128 + hd * HD + nt * 16 + (lane & 15)] = (_Float16)o[r];
      }
    }
    __syncthreads();
  }

  // out-proj: [16x128] x Wo^T + bo -> attnmap (channel-major f32)
  {
    h16 ao[4];
#pragma unroll
    for (int kc = 0; kc < 4; ++kc) ao[kc] = load_a(aO + w * 16 * 128 + kc * 32, 128, lane);
    int gw = blockIdx.x * 8 + w;
    int b = gw >> 10;
    int rem = gw & 1023;
    int wy = rem >> 5, wx = rem & 31;
#pragma unroll
    for (int nt = 0; nt < 8; ++nt) {
      int n0 = nt * 16;
      f8 acc;
      float bv = bo[n0 + (lane & 15)];
#pragma unroll
      for (int r = 0; r < 8; ++r) acc[r] = bv;
#pragma unroll
      for (int ko = 0; ko < 4; ++ko) {
        h16 bfr = load_b_rm(Wo + (size_t)n0 * HID + ko * 32, HID, lane);
        acc = WMMA_F16(ao[ko], bfr, acc);
      }
      int o = n0 + (lane & 15);
      // lane holds tokens t = hi8..hi8+7 -> two window rows of 4 contiguous pixels
      float* dst = attnmap + ((size_t)b * HID + o) * HW + (wy * 4 + (hi8 >> 2)) * Ww + wx * 4;
      float4 v0 = make_float4(acc[0], acc[1], acc[2], acc[3]);
      float4 v1 = make_float4(acc[4], acc[5], acc[6], acc[7]);
      *reinterpret_cast<float4*>(dst) = v0;
      *reinterpret_cast<float4*>(dst + Ww) = v1;
    }
  }
}

// ---------------- group-norm stats: stage 1 partial sums ----------------
__global__ void __launch_bounds__(256) gnstats1_kernel(
    const float* __restrict__ src, float* __restrict__ part) {
  __shared__ float s1[256], s2[256];
  int bg = blockIdx.x / NCHUNK;
  int ck = blockIdx.x % NCHUNK;
  const float4* base = reinterpret_cast<const float4*>(src + (size_t)bg * GELEM + (size_t)ck * CELEM);
  float s = 0.f, q = 0.f;
  for (int i = threadIdx.x; i < CELEM / 4; i += 256) {
    float4 v = base[i];
    s += v.x + v.y + v.z + v.w;
    q += v.x * v.x + v.y * v.y + v.z * v.z + v.w * v.w;
  }
  s1[threadIdx.x] = s; s2[threadIdx.x] = q;
  __syncthreads();
  for (int st = 128; st > 0; st >>= 1) {
    if (threadIdx.x < st) {
      s1[threadIdx.x] += s1[threadIdx.x + st];
      s2[threadIdx.x] += s2[threadIdx.x + st];
    }
    __syncthreads();
  }
  if (threadIdx.x == 0) { part[blockIdx.x * 2] = s1[0]; part[blockIdx.x * 2 + 1] = s2[0]; }
}

// stage 2: finalize mean/rsqrt(var+eps) per (b,group)
__global__ void gnstats2_kernel(const float* __restrict__ part, float* __restrict__ stats) {
  int t = threadIdx.x;
  if (t < Bsz * GROUPS) {
    float s = 0.f, q = 0.f;
    for (int j = 0; j < NCHUNK; ++j) {
      s += part[(t * NCHUNK + j) * 2];
      q += part[(t * NCHUNK + j) * 2 + 1];
    }
    float n = (float)GELEM;
    float mean = s / n;
    float var = q / n - mean * mean;
    stats[t * 2] = mean;
    stats[t * 2 + 1] = rsqrtf(var + EPS);
  }
}

// ---------------- 3) gates GEMM (norm fused into operand) + LSTM elementwise
// M=512(gates) K=128 N=64 spatial columns per WG.
__global__ void __launch_bounds__(256) gates_kernel(
    const float* __restrict__ attn, const float* __restrict__ stats,
    const float* __restrict__ gng, const float* __restrict__ gnb,
    const _Float16* __restrict__ Wg, const float* __restrict__ bg,
    const float* __restrict__ cin,
    float* __restrict__ hout, float* __restrict__ cout) {
  extern __shared__ char smem[];
  _Float16* anT  = (_Float16*)smem;   // [64 p][128 c] normalized attn, f16
  _Float16* gbuf = anT + 64 * 128;    // [64 p][512 o] gates, f16

  int tid = threadIdx.x, lane = tid & 31, wid = tid >> 5;
  int b = blockIdx.x >> 8;
  int p0 = (blockIdx.x & 255) << 6;
  int hi8 = (lane >= 16) ? 8 : 0;

  __builtin_prefetch(Wg + (size_t)wid * 64 * HID, 0, 3);

  // stage normalized input: float4 loads along p, transpose into LDS as f16
#pragma unroll
  for (int i = 0; i < 8; ++i) {
    int lin4 = tid + i * 256;  // 0..2047 == c*16 + plq
    int plq = lin4 & 15;
    int c   = lin4 >> 4;
    int g   = c >> 4;
    float mean = stats[(b * GROUPS + g) * 2];
    float rinv = stats[(b * GROUPS + g) * 2 + 1];
    float gc = gng[c], bc = gnb[c];
    int pl4 = plq << 2;
    float4 v = *reinterpret_cast<const float4*>(attn + ((size_t)b * HID + c) * HW + p0 + pl4);
    anT[(pl4 + 0) * 128 + c] = (_Float16)((v.x - mean) * rinv * gc + bc);
    anT[(pl4 + 1) * 128 + c] = (_Float16)((v.y - mean) * rinv * gc + bc);
    anT[(pl4 + 2) * 128 + c] = (_Float16)((v.z - mean) * rinv * gc + bc);
    anT[(pl4 + 3) * 128 + c] = (_Float16)((v.w - mean) * rinv * gc + bc);
  }
  __syncthreads();

  int m0b = wid * 64;
  f8 acc[4][4];
#pragma unroll
  for (int mt = 0; mt < 4; ++mt) {
    float bb[8];
#pragma unroll
    for (int r = 0; r < 8; ++r) bb[r] = bg[m0b + mt * 16 + r + hi8];
#pragma unroll
    for (int nt = 0; nt < 4; ++nt)
#pragma unroll
      for (int r = 0; r < 8; ++r) acc[mt][nt][r] = bb[r];
  }
#pragma unroll
  for (int ko = 0; ko < 4; ++ko) {
    h16 bfr[4];
#pragma unroll
    for (int nt = 0; nt < 4; ++nt) bfr[nt] = load_b_rm(anT + nt * 16 * 128 + ko * 32, 128, lane);
#pragma unroll
    for (int mt = 0; mt < 4; ++mt) {
      h16 afr = load_a(Wg + (size_t)(m0b + mt * 16) * HID + ko * 32, HID, lane);
#pragma unroll
      for (int nt = 0; nt < 4; ++nt) acc[mt][nt] = WMMA_F16(afr, bfr[nt], acc[mt][nt]);
    }
  }
#pragma unroll
  for (int mt = 0; mt < 4; ++mt)
#pragma unroll
    for (int nt = 0; nt < 4; ++nt)
#pragma unroll
      for (int r = 0; r < 8; ++r)
        gbuf[(nt * 16 + (lane & 15)) * 512 + m0b + mt * 16 + r + hi8] = (_Float16)acc[mt][nt][r];
  __syncthreads();

  // LSTM update: i,f,o,g splits along o; write c' and pre-norm h' straight to d_out
#pragma unroll
  for (int i = 0; i < 8; ++i) {
    int lin4 = tid + i * 256;  // 0..2047 == ch*16 + plq
    int plq = lin4 & 15;
    int ch  = lin4 >> 4;
    int pl4 = plq << 2;
    size_t idx = ((size_t)b * HID + ch) * HW + p0 + pl4;
    float4 cv = *reinterpret_cast<const float4*>(cin + idx);
    float4 cn, hp;
    float cva[4] = {cv.x, cv.y, cv.z, cv.w};
    float cna[4], hpa[4];
#pragma unroll
    for (int j = 0; j < 4; ++j) {
      float gi = (float)gbuf[(pl4 + j) * 512 + ch];
      float gf = (float)gbuf[(pl4 + j) * 512 + 128 + ch];
      float go = (float)gbuf[(pl4 + j) * 512 + 256 + ch];
      float gg = (float)gbuf[(pl4 + j) * 512 + 384 + ch];
      float c1 = sigm(gf) * cva[j] + sigm(gi) * tanhf(gg);
      cna[j] = c1;
      hpa[j] = sigm(go) * tanhf(c1);
    }
    cn = make_float4(cna[0], cna[1], cna[2], cna[3]);
    hp = make_float4(hpa[0], hpa[1], hpa[2], hpa[3]);
    *reinterpret_cast<float4*>(cout + idx) = cn;
    *reinterpret_cast<float4*>(hout + idx) = hp;
  }
}

// ---------------- 4) in-place group-norm apply on h' ----------------
__global__ void __launch_bounds__(256) gnapply_kernel(
    float* __restrict__ data, const float* __restrict__ stats,
    const float* __restrict__ g, const float* __restrict__ bb) {
  size_t base = ((size_t)blockIdx.x * 256 + threadIdx.x) * 4;
  int c = (int)((base >> 14) & 127);  // / HW % HID
  int b = (int)(base >> 21);          // / (HID*HW)
  int gg = c >> 4;
  float mean = stats[(b * GROUPS + gg) * 2];
  float rinv = stats[(b * GROUPS + gg) * 2 + 1];
  float gc = g[c], bc = bb[c];
  float4 v = *reinterpret_cast<float4*>(data + base);
  v.x = (v.x - mean) * rinv * gc + bc;
  v.y = (v.y - mean) * rinv * gc + bc;
  v.z = (v.z - mean) * rinv * gc + bc;
  v.w = (v.w - mean) * rinv * gc + bc;
  *reinterpret_cast<float4*>(data + base) = v;
}

// ---------------- launch ----------------
extern "C" void kernel_launch(void* const* d_in, const int* in_sizes, int n_in,
                              void* d_out, int out_size, void* d_ws, size_t ws_size,
                              hipStream_t stream) {
  (void)in_sizes; (void)n_in; (void)out_size; (void)ws_size;
  const float* x      = (const float*)d_in[0];
  const float* h      = (const float*)d_in[1];
  const float* c      = (const float*)d_in[2];
  const float* Wproj  = (const float*)d_in[3];
  const float* bproj  = (const float*)d_in[4];
  const float* ln_g   = (const float*)d_in[5];
  const float* ln_b   = (const float*)d_in[6];
  const float* Wqkv   = (const float*)d_in[7];
  const float* bqkv   = (const float*)d_in[8];
  const float* Wo     = (const float*)d_in[9];
  const float* bo     = (const float*)d_in[10];
  const float* Wgates = (const float*)d_in[11];
  const float* bgates = (const float*)d_in[12];
  const float* gn_g   = (const float*)d_in[13];
  const float* gn_b   = (const float*)d_in[14];

  char* ws = (char*)d_ws;
  _Float16* projh   = (_Float16*)ws;                       // 33,554,432 B
  float*    attnmap = (float*)(ws + 33554432ull);          // 67,108,864 B
  _Float16* Wp_h    = (_Float16*)(ws + 100663296ull);      //  49,152 B
  _Float16* Wq_h    = (_Float16*)(ws + 100712448ull);      //  98,304 B
  _Float16* Wo_h    = (_Float16*)(ws + 100810752ull);      //  32,768 B
  _Float16* Wg_h    = (_Float16*)(ws + 100843520ull);      // 131,072 B
  float*    part    = (float*)(ws + 100974592ull);         //   8,192 B
  float*    stats1  = (float*)(ws + 100982784ull);         //     512 B
  float*    stats2  = (float*)(ws + 100983296ull);         //     512 B

  float* hout = (float*)d_out;
  float* cout = hout + (size_t)Bsz * HID * HW;

  wcvt_kernel<<<256, 256, 0, stream>>>(Wproj, Wqkv, Wo, Wgates, Wp_h, Wq_h, Wo_h, Wg_h);
  proj_kernel<<<Bsz * (HW / 128), 256, 0, stream>>>(x, h, Wp_h, bproj, projh);
  attn_kernel<<<(Bsz * 1024) / 8, 256, 106496, stream>>>(projh, ln_g, ln_b, Wq_h, bqkv,
                                                         Wo_h, bo, attnmap);
  gnstats1_kernel<<<Bsz * GROUPS * NCHUNK, 256, 0, stream>>>(attnmap, part);
  gnstats2_kernel<<<1, 64, 0, stream>>>(part, stats1);
  gates_kernel<<<Bsz * (HW / 64), 256, 81920, stream>>>(attnmap, stats1, gn_g, gn_b,
                                                        Wg_h, bgates, c, hout, cout);
  gnstats1_kernel<<<Bsz * GROUPS * NCHUNK, 256, 0, stream>>>(hout, part);
  gnstats2_kernel<<<1, 64, 0, stream>>>(part, stats2);
  gnapply_kernel<<<(Bsz * HID * HW) / 1024, 256, 0, stream>>>(hout, stats2, gn_g, gn_b);
}
